// MultiQueryAttention_11381663335090
// MI455X (gfx1250) — compile-verified, matrix operand order checked by reading, not run on hardware
//
#include <hip/hip_runtime.h>
#include <hip/hip_bf16.h>

// ---------------- problem constants ----------------
constexpr int Bc = 2, Tc = 2048, Dc = 1024;
constexpr int Hc = 16, KVHc = 4, HDc = 64;
constexpr int WINDOWc = 256;
constexpr float EPSc = 1e-6f;
constexpr int BTc = Bc * Tc;          // 4096 rows

// ---------------- WMMA types ----------------
typedef __attribute__((ext_vector_type(16))) __bf16 bf16x16;
typedef __attribute__((ext_vector_type(8)))  float  f32x8;
typedef __attribute__((ext_vector_type(4)))  unsigned int u32x4;
typedef int i32x4v __attribute__((__vector_size__(16)));   // matches builtin param

union FragU { bf16x16 v; u32x4 q[2]; };

__device__ __forceinline__ f32x8 wmma_bf16(bf16x16 a, bf16x16 b, f32x8 c) {
    return __builtin_amdgcn_wmma_f32_16x16x32_bf16(false, a, false, b, (short)0, c, false, false);
}

__device__ __forceinline__ unsigned short f2bf(float f) {
    unsigned int x = __float_as_uint(f);
    unsigned int r = (x + 0x7FFFu + ((x >> 16) & 1u)) >> 16;
    return (unsigned short)r;
}

// ---------------- gfx1250 async global->LDS staging ----------------
#define GLOBAL_AS_Q __attribute__((address_space(1)))
#define LDS_AS_Q    __attribute__((address_space(3)))

#if __has_builtin(__builtin_amdgcn_global_load_async_to_lds_b128)
#define HAVE_ASYNC_LDS 1
#endif

__device__ __forceinline__ void stage16(const unsigned short* g, unsigned short* l) {
#ifdef HAVE_ASYNC_LDS
    // emits global_load_async_to_lds_b128 (ASYNCcnt-tracked DMA into LDS)
    __builtin_amdgcn_global_load_async_to_lds_b128(
        (GLOBAL_AS_Q i32x4v*)g, (LDS_AS_Q i32x4v*)l, 0, 0);
#else
    *reinterpret_cast<u32x4*>(l) = *reinterpret_cast<const u32x4*>(g);
#endif
}

__device__ __forceinline__ void wait_async() {
#ifdef HAVE_ASYNC_LDS
#if __has_builtin(__builtin_amdgcn_s_wait_asynccnt)
    __builtin_amdgcn_s_wait_asynccnt(0);
#else
    asm volatile("s_wait_asynccnt 0x0" ::: "memory");
#endif
#endif
}

// Load a 16x32 bf16 A-style fragment from a row-major matrix (ld = row stride
// in elements). Also valid as a B fragment of W^T when W is row-major [N,K].
// Lane l<16 : row=row0+l,   K = {k0..k0+7, k0+16..k0+23}
// Lane l>=16: row=row0+l-16,K = {k0+8..k0+15, k0+24..k0+31}
__device__ __forceinline__ bf16x16 load_frag(const unsigned short* base, int row0,
                                             int ld, int k0, int lane) {
    int r = lane & 15;
    int c = k0 + ((lane >> 4) << 3);
    const unsigned short* p = base + (size_t)(row0 + r) * ld + c;
    FragU u;
    u.q[0] = *reinterpret_cast<const u32x4*>(p);
    u.q[1] = *reinterpret_cast<const u32x4*>(p + 16);
    return u.v;
}

// Same fragment indexing, reading a 64x32 row-major bf16 panel held in LDS.
__device__ __forceinline__ bf16x16 lds_frag(const unsigned short* panel, int row0, int lane) {
    int r = lane & 15;
    int c = (lane >> 4) << 3;
    const unsigned short* p = panel + (row0 + r) * 32 + c;
    FragU u;
    u.q[0] = *reinterpret_cast<const u32x4*>(p);
    u.q[1] = *reinterpret_cast<const u32x4*>(p + 16);
    return u.v;
}

// ---------------- prep kernels ----------------
__global__ void cvt_f32_to_bf16(const float* __restrict__ src,
                                unsigned short* __restrict__ dst, size_t n) {
    size_t i = (size_t)blockIdx.x * blockDim.x + threadIdx.x;
    if (i < n) dst[i] = f2bf(src[i]);
}

__global__ void add_ve_cvt(const float* __restrict__ x, const float* __restrict__ ve,
                           unsigned short* __restrict__ dst, size_t n, size_t td) {
    size_t i = (size_t)blockIdx.x * blockDim.x + threadIdx.x;
    if (i < n) dst[i] = f2bf(x[i] + ve[i % td]);
}

// ------- cooperative WMMA GEMM: C[M,N] f32 = A[M,K]bf16 * W[N,K]bf16^T -------
// Block = 256 threads = 8 waves. All 8 waves share one 64-column W strip whose
// 64x32 k-panels are staged into LDS (double-buffered, async global->LDS DMA).
// Wave w computes rows [ (bm*8+w)*16 , +16 ). Requires M % 128 == 0, N % 64 == 0.
__global__ void gemm_bf16_wmma(const unsigned short* __restrict__ A,
                               const unsigned short* __restrict__ W,
                               float* __restrict__ C, int M, int N, int K) {
    __shared__ __align__(16) unsigned short wpanel[2][64 * 32];
    int ntiles = N >> 6;
    int bm = blockIdx.x / ntiles;
    int nt = blockIdx.x % ntiles;
    int waveid = threadIdx.x >> 5;
    int lane   = threadIdx.x & 31;
    int mt = bm * 8 + waveid;

    // staging assignment: thread -> one 16B chunk of the 64x32 panel
    int srow   = threadIdx.x >> 2;         // 0..63
    int schunk = (threadIdx.x & 3) << 3;   // 0,8,16,24 (elements)
    const unsigned short* wsrc = W + (size_t)(nt * 64 + srow) * K + schunk;

    int nk = K >> 5;
    stage16(wsrc, &wpanel[0][srow * 32 + schunk]);

    f32x8 acc0 = {}, acc1 = {}, acc2 = {}, acc3 = {};
    for (int ki = 0; ki < nk; ++ki) {
        int k0 = ki << 5;
        int buf = ki & 1;
        wait_async();
        __syncthreads();                    // panel[buf] ready for all waves
        if (ki + 1 < nk)                    // start DMA of next panel
            stage16(wsrc + k0 + 32, &wpanel[buf ^ 1][srow * 32 + schunk]);

        bf16x16 a = load_frag(A, mt * 16, K, k0, lane);
        if (ki + 1 < nk)
            __builtin_prefetch(A + (size_t)(mt * 16 + (lane & 15)) * K + k0 + 32, 0, 1);

        const unsigned short* pan = wpanel[buf];
        bf16x16 b0 = lds_frag(pan,  0, lane);
        bf16x16 b1 = lds_frag(pan, 16, lane);
        bf16x16 b2 = lds_frag(pan, 32, lane);
        bf16x16 b3 = lds_frag(pan, 48, lane);
        acc0 = wmma_bf16(a, b0, acc0);
        acc1 = wmma_bf16(a, b1, acc1);
        acc2 = wmma_bf16(a, b2, acc2);
        acc3 = wmma_bf16(a, b3, acc3);
        __syncthreads();                    // all reads done before re-stage
    }
    int hi = lane >> 4, col = lane & 15;
    #pragma unroll
    for (int r = 0; r < 8; ++r) {
        size_t row = (size_t)(mt * 16 + r + 8 * hi);
        float* crow = C + row * N + nt * 64;
        crow[ 0 + col] = acc0[r];
        crow[16 + col] = acc1[r];
        crow[32 + col] = acc2[r];
        crow[48 + col] = acc3[r];
    }
}

// ---------------- RoPE + RMS-norm, fp32 in -> bf16 out --------------------
// qf: [b,t,H,64], kf: [b,t,KVH,64]  ->  qh: [b,H,t,64], kh: [b,KVH,t,64]
__global__ void rope_rms(const float* __restrict__ qf, const float* __restrict__ kf,
                         const float* __restrict__ cosb, const float* __restrict__ sinb,
                         unsigned short* __restrict__ qh, unsigned short* __restrict__ kh) {
    int w    = (int)(((size_t)blockIdx.x * blockDim.x + threadIdx.x) >> 5);
    int lane = threadIdx.x & 31;
    int nq = Bc * Tc * Hc;
    int total = nq + Bc * Tc * KVHc;
    if (w >= total) return;

    const float* src;
    unsigned short* dst;
    int t;
    if (w < nq) {
        int head = w % Hc; t = (w / Hc) % Tc; int b = w / (Hc * Tc);
        src = qf + ((size_t)(b * Tc + t) * (Hc * HDc) + head * HDc);
        dst = qh + (((size_t)b * Hc + head) * Tc + t) * HDc;
    } else {
        int idx = w - nq;
        int head = idx % KVHc; t = (idx / KVHc) % Tc; int b = idx / (KVHc * Tc);
        src = kf + ((size_t)(b * Tc + t) * (KVHc * HDc) + head * HDc);
        dst = kh + (((size_t)b * KVHc + head) * Tc + t) * HDc;
    }
    float xr = src[2 * lane], xi = src[2 * lane + 1];
    float c = cosb[t * 32 + lane], s = sinb[t * 32 + lane];
    float orr = xr * c - xi * s;
    float oii = xr * s + xi * c;
    float ss = orr * orr + oii * oii;
    #pragma unroll
    for (int m = 1; m < 32; m <<= 1) ss += __shfl_xor(ss, m, 32);
    float inv = rsqrtf(ss * (1.0f / HDc) + EPSc);
    orr *= inv; oii *= inv;
    dst[2 * lane]     = f2bf(orr);
    dst[2 * lane + 1] = f2bf(oii);
}

// vf: [b,t,KVH,64] f32  ->  vt: [b,KVH,64,T] bf16
__global__ void v_transpose_cvt(const float* __restrict__ vf,
                                unsigned short* __restrict__ vt) {
    size_t i = (size_t)blockIdx.x * blockDim.x + threadIdx.x;
    size_t n = (size_t)BTc * KVHc * HDc;
    if (i >= n) return;
    int d  = (int)(i % HDc);
    size_t r1 = i / HDc;
    int kv = (int)(r1 % KVHc);
    size_t r2 = r1 / KVHc;
    int t  = (int)(r2 % Tc);
    int b  = (int)(r2 / Tc);
    vt[(((size_t)b * KVHc + kv) * HDc + d) * Tc + t] = f2bf(vf[i]);
}

// ---------------- sliding-window attention (flash-style, WMMA) -------------
// One wave per (b, h, 16-query tile). Key pairs of 16-tiles => WMMA K=32.
__global__ void swa_attn(const unsigned short* __restrict__ qh,
                         const unsigned short* __restrict__ kh,
                         const unsigned short* __restrict__ vt,
                         unsigned short* __restrict__ yh) {
    __shared__ __align__(16) unsigned short pl_s[4][16 * 32]; // per-wave P staging
    int w    = (int)(((size_t)blockIdx.x * blockDim.x + threadIdx.x) >> 5);
    int lane = threadIdx.x & 31;
    int wslot = threadIdx.x >> 5;

    int qtiles = Tc / 16;
    int total = Bc * Hc * qtiles;
    if (w >= total) return;
    int qt = w % qtiles;
    int h  = (w / qtiles) % Hc;
    int b  = w / (qtiles * Hc);
    int kv = h / (Hc / KVHc);

    const unsigned short* qbase = qh + ((size_t)(b * Hc + h) * Tc) * HDc;
    const unsigned short* kbase = kh + ((size_t)(b * KVHc + kv) * Tc) * HDc;
    const unsigned short* vbase = vt + ((size_t)(b * KVHc + kv) * HDc) * Tc;

    bf16x16 aq0 = load_frag(qbase, qt * 16, HDc, 0,  lane);
    bf16x16 aq1 = load_frag(qbase, qt * 16, HDc, 32, lane);

    f32x8 O0 = {}, O1 = {}, O2 = {}, O3 = {};
    float m_run[8], l_run[8];
    #pragma unroll
    for (int r = 0; r < 8; ++r) { m_run[r] = -1e30f; l_run[r] = 0.0f; }

    int hi = lane >> 4, col = lane & 15;
    unsigned short* pl = pl_s[wslot];

    // window covers key tiles [qt-16, qt]; pad to 18 tiles starting at qt-17
    for (int pt = 0; pt < 9; ++pt) {
        int kt0 = qt - 17 + 2 * pt;
        float Sv[2][8];
        int ktc[2];
        #pragma unroll
        for (int tt = 0; tt < 2; ++tt) {
            int kt = kt0 + tt;
            ktc[tt] = kt < 0 ? 0 : kt;
            f32x8 sc = {};
            bf16x16 bk0 = load_frag(kbase, ktc[tt] * 16, HDc, 0,  lane);
            bf16x16 bk1 = load_frag(kbase, ktc[tt] * 16, HDc, 32, lane);
            sc = wmma_bf16(aq0, bk0, sc);
            sc = wmma_bf16(aq1, bk1, sc);
            #pragma unroll
            for (int r = 0; r < 8; ++r) {
                int qg = qt * 16 + r + 8 * hi;
                int kg = kt * 16 + col;
                int dist = qg - kg;
                bool ok = (kg >= 0) && (dist >= 0) && (dist < WINDOWc);
                Sv[tt][r] = ok ? sc[r] * 0.125f : -1e38f;   // 1/sqrt(64)
            }
        }
        // row-wise max across the 16 columns (lanes of same half)
        float tm[8];
        #pragma unroll
        for (int r = 0; r < 8; ++r) tm[r] = fmaxf(Sv[0][r], Sv[1][r]);
        #pragma unroll
        for (int m = 1; m < 16; m <<= 1)
            #pragma unroll
            for (int r = 0; r < 8; ++r) tm[r] = fmaxf(tm[r], __shfl_xor(tm[r], m, 32));
        float alpha[8];
        #pragma unroll
        for (int r = 0; r < 8; ++r) {
            float mn = fmaxf(m_run[r], tm[r]);
            alpha[r] = __expf(m_run[r] - mn);
            m_run[r] = mn;
        }
        float ps[8];
        #pragma unroll
        for (int r = 0; r < 8; ++r) {
            float p0 = __expf(Sv[0][r] - m_run[r]);
            float p1 = __expf(Sv[1][r] - m_run[r]);
            Sv[0][r] = p0; Sv[1][r] = p1;
            ps[r] = p0 + p1;
        }
        #pragma unroll
        for (int m = 1; m < 16; m <<= 1)
            #pragma unroll
            for (int r = 0; r < 8; ++r) ps[r] += __shfl_xor(ps[r], m, 32);
        #pragma unroll
        for (int r = 0; r < 8; ++r) {
            l_run[r] = l_run[r] * alpha[r] + ps[r];
            O0[r] *= alpha[r]; O1[r] *= alpha[r];
            O2[r] *= alpha[r]; O3[r] *= alpha[r];
        }
        // stage P (C-layout) through LDS, reload in A-layout (single-wave,
        // DS pipeline is in-order for this wave -> no barrier needed)
        #pragma unroll
        for (int tt = 0; tt < 2; ++tt)
            #pragma unroll
            for (int r = 0; r < 8; ++r)
                pl[(r + 8 * hi) * 32 + tt * 16 + col] = f2bf(Sv[tt][r]);
        FragU up;
        {
            int rr = col, cc = hi * 8;
            up.q[0] = *reinterpret_cast<const u32x4*>(&pl[rr * 32 + cc]);
            up.q[1] = *reinterpret_cast<const u32x4*>(&pl[rr * 32 + cc + 16]);
        }
        // P(16x32) @ V(32x64): B fragments contiguous along keys in vt
        #pragma unroll
        for (int j = 0; j < 4; ++j) {
            const unsigned short* vrow = vbase + (size_t)(j * 16 + col) * Tc;
            FragU uv;
            uv.q[0] = *reinterpret_cast<const u32x4*>(&vrow[ktc[0] * 16 + hi * 8]);
            uv.q[1] = *reinterpret_cast<const u32x4*>(&vrow[ktc[1] * 16 + hi * 8]);
            f32x8* Oj = (j == 0) ? &O0 : (j == 1) ? &O1 : (j == 2) ? &O2 : &O3;
            *Oj = wmma_bf16(up.v, uv.v, *Oj);
        }
    }
    // normalize and store y as bf16 [b, t, h*64 + d]
    #pragma unroll
    for (int r = 0; r < 8; ++r) {
        float inv = 1.0f / l_run[r];
        size_t row = (size_t)b * Tc + qt * 16 + r + 8 * hi;
        unsigned short* yrow = yh + row * (Hc * HDc) + h * HDc;
        yrow[ 0 + col] = f2bf(O0[r] * inv);
        yrow[16 + col] = f2bf(O1[r] * inv);
        yrow[32 + col] = f2bf(O2[r] * inv);
        yrow[48 + col] = f2bf(O3[r] * inv);
    }
}

// ---------------- host launcher ----------------
extern "C" void kernel_launch(void* const* d_in, const int* in_sizes, int n_in,
                              void* d_out, int out_size, void* d_ws, size_t ws_size,
                              hipStream_t stream) {
    const float* x    = (const float*)d_in[0];
    const float* cosb = (const float*)d_in[1];
    const float* sinb = (const float*)d_in[2];
    const float* wq   = (const float*)d_in[3];
    const float* wk   = (const float*)d_in[4];
    const float* wv   = (const float*)d_in[5];
    const float* wo   = (const float*)d_in[6];
    const float* ve   = (const float*)d_in[7];
    float* out = (float*)d_out;

    char* ws = (char*)d_ws;
    size_t off = 0;
    auto alloc = [&](size_t bytes) -> void* {
        void* p = ws + off;
        off += (bytes + 255) & ~(size_t)255;
        return p;
    };
    unsigned short* xh   = (unsigned short*)alloc((size_t)BTc * Dc * 2);
    unsigned short* vinh = (unsigned short*)alloc((size_t)BTc * Dc * 2);
    unsigned short* wqh  = (unsigned short*)alloc((size_t)Dc * Dc * 2);
    unsigned short* wkh  = (unsigned short*)alloc((size_t)KVHc * HDc * Dc * 2);
    unsigned short* wvh  = (unsigned short*)alloc((size_t)KVHc * HDc * Dc * 2);
    unsigned short* woh  = (unsigned short*)alloc((size_t)Dc * Dc * 2);
    float* qf            = (float*)alloc((size_t)BTc * Hc * HDc * 4);
    float* kf            = (float*)alloc((size_t)BTc * KVHc * HDc * 4);
    float* vf            = (float*)alloc((size_t)BTc * KVHc * HDc * 4);
    unsigned short* qhb  = (unsigned short*)alloc((size_t)BTc * Hc * HDc * 2);
    unsigned short* khb  = (unsigned short*)alloc((size_t)BTc * KVHc * HDc * 2);
    unsigned short* vtb  = (unsigned short*)alloc((size_t)BTc * KVHc * HDc * 2);
    unsigned short* yhb  = (unsigned short*)alloc((size_t)BTc * Hc * HDc * 2);

    auto blocks1d = [](size_t n, int bs) { return (unsigned)((n + bs - 1) / bs); };

    // 1) fp32 -> bf16 conversions
    size_t nXD = (size_t)BTc * Dc;
    cvt_f32_to_bf16<<<blocks1d(nXD, 256), 256, 0, stream>>>(x, xh, nXD);
    add_ve_cvt<<<blocks1d(nXD, 256), 256, 0, stream>>>(x, ve, vinh, nXD, (size_t)Tc * Dc);
    cvt_f32_to_bf16<<<blocks1d((size_t)Dc * Dc, 256), 256, 0, stream>>>(wq, wqh, (size_t)Dc * Dc);
    cvt_f32_to_bf16<<<blocks1d((size_t)KVHc * HDc * Dc, 256), 256, 0, stream>>>(wk, wkh, (size_t)KVHc * HDc * Dc);
    cvt_f32_to_bf16<<<blocks1d((size_t)KVHc * HDc * Dc, 256), 256, 0, stream>>>(wv, wvh, (size_t)KVHc * HDc * Dc);
    cvt_f32_to_bf16<<<blocks1d((size_t)Dc * Dc, 256), 256, 0, stream>>>(wo, woh, (size_t)Dc * Dc);

    // 2) projection GEMMs (cooperative WMMA, LDS-staged weights)
    auto gemm_launch = [&](const unsigned short* A, const unsigned short* W,
                           float* C, int M, int N, int K) {
        unsigned blocks = (unsigned)((M / 128) * (N / 64));
        gemm_bf16_wmma<<<blocks, 256, 0, stream>>>(A, W, C, M, N, K);
    };
    gemm_launch(xh,   wqh, qf, BTc, Hc * HDc,   Dc);   // Q: 4096x1024
    gemm_launch(xh,   wkh, kf, BTc, KVHc * HDc, Dc);   // K: 4096x256
    gemm_launch(vinh, wvh, vf, BTc, KVHc * HDc, Dc);   // V: 4096x256

    // 3) RoPE + RMS-norm -> bf16, head-major layouts
    {
        size_t waves = (size_t)BTc * (Hc + KVHc);
        rope_rms<<<blocks1d(waves * 32, 256), 256, 0, stream>>>(qf, kf, cosb, sinb, qhb, khb);
    }
    // 4) V -> transposed bf16 [b,kvh,64,T]
    {
        size_t n = (size_t)BTc * KVHc * HDc;
        v_transpose_cvt<<<blocks1d(n, 256), 256, 0, stream>>>(vf, vtb);
    }
    // 5) windowed attention
    {
        size_t waves = (size_t)Bc * Hc * (Tc / 16);
        swa_attn<<<blocks1d(waves * 32, 128), 128, 0, stream>>>(qhb, khb, vtb, yhb);
    }
    // 6) output projection -> fp32 d_out
    gemm_launch(yhb, woh, out, BTc, Dc, Hc * HDc);
}
